// BertEmbeddings_65446711657067
// MI455X (gfx1250) — compile-verified, matrix operand order checked by reading, not run on hardware
//
#include <hip/hip_runtime.h>
#include <hip/hip_bf16.h>

// Problem constants (match reference)
#define BB 32
#define SS 512
#define HH 768
#define NC 32
#define HL 16
#define LN_EPS 1e-12f

typedef __attribute__((ext_vector_type(16))) _Float16 v16h;
typedef __attribute__((ext_vector_type(8)))  float    v8f;

// ---------------------------------------------------------------------------
// Kernel 1: variable-length mean-pool of header token embeddings via WMMA.
// One wave per (b,c) pair.  pooled[b,c,h] = (1/max(len,1)) * sum_{l<len} word_emb[hid[l]][h]
//
// WMMA trick: A has only row 0 populated (all ones; row M=0 lives in lanes 0
// and 16 per the documented 16-bit A layout).  B holds mask-weighted header
// rows, one column per lane (n = lane & 15); lanes 16-31 contribute zeros.
// D[0,n] = sum over all K slots of column n, which is invariant to the exact
// K->slot permutation of the B layout.
//
// Branch-free gather: every lane always loads a valid row and multiplies by a
// 0/1 mask weight, so the 16 gather loads issue as one unpredicated batch
// instead of 16 exec-masked load+wait sequences.
// ---------------------------------------------------------------------------
__global__ __launch_bounds__(128)
void pool_headers_wmma(const int* __restrict__ header_ids,   // [B,NC,HL]
                       const int* __restrict__ header_len,   // [B,NC]
                       const float* __restrict__ word_emb,   // [VOCAB,H]
                       float* __restrict__ pooled)           // [B,NC,H] (ws)
{
    const int lane = threadIdx.x & 31;
    const int wid  = __builtin_amdgcn_readfirstlane((int)(threadIdx.x >> 5));
    const int pair = blockIdx.x * 4 + wid;           // b*NC + c, uniform per wave

    const int len       = header_len[pair];
    const int* hid      = header_ids + (size_t)pair * HL;
    const float inv_den = 1.0f / (float)(len > 0 ? len : 1);

    // A matrix (16x32 f16): row M=0 -> lanes 0 and 16 hold all-ones.
    v16h a = {};
    if ((lane & 15) == 0) {
#pragma unroll
        for (int i = 0; i < 16; ++i) a[i] = (_Float16)1.0f;
    }

    const int  col = lane & 15;
    const bool low = lane < 16;

    // Per-header-token base pointer (always valid) + 0/1 mask weight.
    const float* rptr[HL];
    float        wgt[HL];
#pragma unroll
    for (int l = 0; l < HL; ++l) {
        rptr[l] = word_emb + (size_t)hid[l] * HH + col;
        wgt[l]  = (low && l < len) ? 1.0f : 0.0f;
    }

    for (int t = 0; t < HH / 16; ++t) {
        const int off = t * 16;
        v16h bm;
#pragma unroll
        for (int l = 0; l < HL; ++l) {
            bm[l] = (_Float16)(rptr[l][off] * wgt[l]);
        }
        v8f acc = {};
        // D = A x B + C  (emits v_wmma_f32_16x16x32_f16)
        acc = __builtin_amdgcn_wmma_f32_16x16x32_f16(
                  false, a, false, bm, (short)0, acc, false, false);
        if (low)
            pooled[(size_t)pair * HH + off + col] = acc[0] * inv_den;
    }
}

// ---------------------------------------------------------------------------
// Kernel 2: fused embedding sum + scatter-override + TF LayerNorm.
// One wave32 per token row (H=768 -> 6 float4 per lane, fully coalesced).
// 8 waves (256 threads) per block.  Bandwidth-bound: ~105 MB of HBM traffic
// total -> ~5 us at 23.3 TB/s.
// ---------------------------------------------------------------------------
__global__ __launch_bounds__(256)
void embed_layernorm(const int* __restrict__ input_ids,       // [B,S]
                     const int* __restrict__ token_type_ids,  // [B,S]
                     const int* __restrict__ match_type_ids,  // [B,S]
                     const int* __restrict__ type_idx,        // [B,S]
                     const int* __restrict__ col_pos,         // [B,NC]
                     const int* __restrict__ col_idx,         // [B,NC]
                     const int* __restrict__ header_len,      // [B,NC]
                     const float* __restrict__ word_emb,      // [VOCAB,H]
                     const float* __restrict__ pos_emb,       // [S,H]
                     const float* __restrict__ tok_type_emb,  // [2,H]
                     const float* __restrict__ match_emb,     // [11,H]
                     const float* __restrict__ type_emb,      // [6,H]
                     const float* __restrict__ ln_w,          // [H]
                     const float* __restrict__ ln_b,          // [H]
                     const float* __restrict__ pooled,        // [B,NC,H] (ws)
                     float* __restrict__ out)                 // [B,S,H]
{
    const int lane  = threadIdx.x & 31;
    const int wid   = __builtin_amdgcn_readfirstlane((int)(threadIdx.x >> 5));
    const int token = blockIdx.x * 8 + wid;          // uniform per wave
    const int b     = token >> 9;                    // S = 512
    const int s     = token & (SS - 1);

    // Scatter-override detection: is s one of the NC column positions?
    // All addresses are wave-uniform -> SALU/SMEM path.
    int ov_src = -1;
    for (int c = 0; c < NC; ++c) {
        if (col_pos[b * NC + c] == s) {
            int ci = col_idx[b * NC + c];
            if (header_len[b * NC + ci] > 0) ov_src = ci;
        }
    }

    const size_t tokoff = (size_t)b * SS + s;
    const int wrow = input_ids[tokoff];
    const int tt   = token_type_ids[tokoff];
    const int mt   = match_type_ids[tokoff];
    const int ti   = type_idx[tokoff];

    const float4* base = (ov_src >= 0)
        ? (const float4*)(pooled   + ((size_t)b * NC + ov_src) * HH)
        : (const float4*)(word_emb + (size_t)wrow * HH);
    const float4* prow = (const float4*)(pos_emb      + (size_t)s  * HH);
    const float4* trow = (const float4*)(tok_type_emb + (size_t)tt * HH);
    const float4* mrow = (const float4*)(match_emb    + (size_t)mt * HH);
    const float4* yrow = (const float4*)(type_emb     + (size_t)ti * HH);

    float4 x[6];
    float sum = 0.0f, sq = 0.0f;
#pragma unroll
    for (int v = 0; v < 6; ++v) {
        const int i = v * 32 + lane;                 // 192 float4 per row
        float4 a = base[i];
        float4 p = prow[i];
        float4 t = trow[i];
        float4 m = mrow[i];
        float4 y = yrow[i];
        float4 e;
        e.x = a.x + p.x + t.x + m.x + y.x;
        e.y = a.y + p.y + t.y + m.y + y.y;
        e.z = a.z + p.z + t.z + m.z + y.z;
        e.w = a.w + p.w + t.w + m.w + y.w;
        x[v] = e;
        sum += e.x + e.y + e.z + e.w;
        sq  += e.x * e.x + e.y * e.y + e.z * e.z + e.w * e.w;
    }

    // wave32 butterfly reduction
#pragma unroll
    for (int off = 16; off > 0; off >>= 1) {
        sum += __shfl_xor(sum, off, 32);
        sq  += __shfl_xor(sq,  off, 32);
    }

    const float mean = sum * (1.0f / HH);
    float var = sq * (1.0f / HH) - mean * mean;
    var = var < 0.0f ? 0.0f : var;
    const float rstd = rsqrtf(var + LN_EPS);

    float4* o = (float4*)(out + tokoff * HH);
    const float4* wv = (const float4*)ln_w;
    const float4* bv = (const float4*)ln_b;
#pragma unroll
    for (int v = 0; v < 6; ++v) {
        const int i = v * 32 + lane;
        float4 e = x[v];
        float4 w = wv[i];
        float4 g = bv[i];
        float4 r;
        r.x = w.x * ((e.x - mean) * rstd) + g.x;
        r.y = w.y * ((e.y - mean) * rstd) + g.y;
        r.z = w.z * ((e.z - mean) * rstd) + g.z;
        r.w = w.w * ((e.w - mean) * rstd) + g.w;
        o[i] = r;
    }
}

// ---------------------------------------------------------------------------
// Launch: kernel 1 writes pooled rows into d_ws (needs B*NC*H*4 = 3 MB),
// kernel 2 consumes them.  Same stream => ordered.
// ---------------------------------------------------------------------------
extern "C" void kernel_launch(void* const* d_in, const int* in_sizes, int n_in,
                              void* d_out, int out_size, void* d_ws, size_t ws_size,
                              hipStream_t stream) {
    const int*   input_ids      = (const int*)d_in[0];
    const int*   header_ids     = (const int*)d_in[1];
    const int*   token_type_ids = (const int*)d_in[2];
    const int*   match_type_ids = (const int*)d_in[3];
    const int*   type_idx       = (const int*)d_in[4];
    const int*   col_pos        = (const int*)d_in[5];
    const int*   col_idx        = (const int*)d_in[6];
    const int*   header_len     = (const int*)d_in[7];
    const float* word_emb       = (const float*)d_in[8];
    const float* pos_emb        = (const float*)d_in[9];
    const float* tok_type_emb   = (const float*)d_in[10];
    const float* match_emb      = (const float*)d_in[11];
    const float* type_emb       = (const float*)d_in[12];
    const float* ln_w           = (const float*)d_in[13];
    const float* ln_b           = (const float*)d_in[14];
    float*       out            = (float*)d_out;
    float*       pooled         = (float*)d_ws;   // B*NC*H f32 = 3 MB

    // Kernel 1: one wave per (b,c); 4 waves per 128-thread block.
    pool_headers_wmma<<<(BB * NC) / 4, 128, 0, stream>>>(
        header_ids, header_len, word_emb, pooled);

    // Kernel 2: one wave per token; 8 waves per 256-thread block.
    embed_layernorm<<<(BB * SS) / 8, 256, 0, stream>>>(
        input_ids, token_type_ids, match_type_ids, type_idx,
        col_pos, col_idx, header_len,
        word_emb, pos_emb, tok_type_emb, match_emb, type_emb,
        ln_w, ln_b, pooled, out);
}
